// GRU4REC_11613591568397
// MI455X (gfx1250) — matrix operand order repaired and checked by reading, not compiled, not run on hardware
//
#include <hip/hip_runtime.h>
#include <hip/hip_bf16.h>
#include <math.h>

typedef __attribute__((ext_vector_type(16))) __bf16 v16bf;
typedef __attribute__((ext_vector_type(8)))  __bf16 v8bf;
typedef __attribute__((ext_vector_type(8)))  float  v8f;

__device__ __forceinline__ unsigned short f2bf(float f) {
  union { float f; unsigned int u; } v; v.f = f;
  unsigned int u = v.u;
  u = (u + 0x7FFFu + ((u >> 16) & 1u)) >> 16;   // round-to-nearest-even
  return (unsigned short)u;
}

// ---------------------------------------------------------------------------
// bf16 WMMA GEMM, register-blocked 4x4 tiles per wave (64x64 f32 output).
// C[M,N] = A[M,K] * W[N,K]^T (+ bias[N]); A/W bf16 row-major, C f32.
// K is a template constant so the 4 A-row offsets (mi*16*K elems) fold into
// the 24-bit global_load immediate -> single A base address, no spills.
// __launch_bounds__(128) + waves_per_eu(1) give the ~224-VGPR live set
// (16 f32 accumulators + one K-chunk of operands) room without scratch spill.
// Per K-chunk: 4 A loads + 4 B loads feed 16 v_wmma_f32_16x16x32_bf16.
// Per-lane operand layouts used here (contiguous k, same order in A and B):
//   A (16x32): lane l: row m=l&15, half=l>>4; elems A[m][k0+half*16..+15]
//   B (32x16): lane l: col n=l&15, half=l>>4; elems W[n][k0+half*16..+15]
//   C/D:       lane l: col=l&15, rows=(l>>4)*8 + vgpr(0..7)
// N tail handled by clamping tile index (boundary tiles recomputed with
// identical values -> deterministic, race-free by value).
// ---------------------------------------------------------------------------
template <int K>
__global__ __launch_bounds__(128)
__attribute__((amdgpu_waves_per_eu(1)))
void gemm_bf16_wmma4x4(const unsigned short* __restrict__ A,
                       const unsigned short* __restrict__ W,
                       const float* __restrict__ bias,
                       float* __restrict__ C,
                       int Ntiles, long ldc)
{
  const int lane   = threadIdx.x & 31;
  const int wave   = threadIdx.x >> 5;
  const int tnBase = (blockIdx.x * (blockDim.x >> 5) + wave) * 4;
  const int tmBase = blockIdx.y * 4;
  if (tnBase >= Ntiles) return;       // wave-uniform exit: EXEC all-1 for WMMA

  const int half = lane >> 4;
  const int r    = lane & 15;

  int tnc[4];
  #pragma unroll
  for (int ni = 0; ni < 4; ++ni) {
    int tn = tnBase + ni;
    tnc[ni] = (tn < Ntiles) ? tn : (Ntiles - 1);
  }

  // Single A base; mi tiles live at compile-time offsets mi*16*K elements.
  const unsigned short* Ab = A + (size_t)(tmBase * 16 + r) * (size_t)K + half * 16;
  const unsigned short* Wp[4];
  #pragma unroll
  for (int ni = 0; ni < 4; ++ni)
    Wp[ni] = W + (size_t)(tnc[ni] * 16 + r) * (size_t)K + half * 16;

  v8f acc[16];
  #pragma unroll
  for (int i = 0; i < 16; ++i) acc[i] = {};

  #pragma unroll 1
  for (int k0 = 0; k0 < K; k0 += 32) {
    v16bf a[4], b[4];
    #pragma unroll
    for (int mi = 0; mi < 4; ++mi) {
      v8bf a0 = *(const v8bf*)(Ab + mi * 16 * K + k0);
      v8bf a1 = *(const v8bf*)(Ab + mi * 16 * K + k0 + 8);
      a[mi] = __builtin_shufflevector(a0, a1, 0,1,2,3,4,5,6,7,8,9,10,11,12,13,14,15);
    }
    #pragma unroll
    for (int ni = 0; ni < 4; ++ni)
      b[ni] = *(const v16bf*)(Wp[ni] + k0);
    #pragma unroll
    for (int mi = 0; mi < 4; ++mi)
      #pragma unroll
      for (int ni = 0; ni < 4; ++ni)
        acc[mi * 4 + ni] = __builtin_amdgcn_wmma_f32_16x16x32_bf16(
            false, a[mi], false, b[ni], (short)0, acc[mi * 4 + ni], false, false);
  }

  #pragma unroll
  for (int ni = 0; ni < 4; ++ni) {
    const long colg = (long)tnc[ni] * 16 + r;
    const float bv  = bias ? bias[colg] : 0.0f;
    #pragma unroll
    for (int mi = 0; mi < 4; ++mi) {
      const long rowbase = (long)(tmBase + mi) * 16 + (long)half * 8;
      #pragma unroll
      for (int i = 0; i < 8; ++i)
        C[(rowbase + i) * ldc + colg] = acc[mi * 4 + ni][i] + bv;
    }
  }
}

// f32 -> bf16 bulk convert
__global__ void cvt_f32_bf16(const float* __restrict__ src,
                             unsigned short* __restrict__ dst, long n)
{
  long i = (long)blockIdx.x * blockDim.x + threadIdx.x;
  if (i < n) dst[i] = f2bf(src[i]);
}

// Per-timestep operand prep: Xb = bf16(emb[idx[:,t]]), Hb = bf16(h).
__global__ void gru_prep(const int* __restrict__ idx, int t, int T,
                         const unsigned short* __restrict__ embb,
                         const float* __restrict__ h,
                         unsigned short* __restrict__ Xb,
                         unsigned short* __restrict__ Hb, int rows)
{
  int g = blockIdx.x * blockDim.x + threadIdx.x;
  if (g >= rows * 256) return;
  int i = g >> 8, j = g & 255;
  Xb[g] = embb[(size_t)idx[i * T + t] * 256 + j];
  Hb[g] = f2bf(h[g]);
}

// GRU gate math + in-flight length gather (out = h_t * mask[:,t] at t==len[i]).
__global__ void gru_gate(const float* __restrict__ GI,
                         const float* __restrict__ GH,
                         const float* __restrict__ bih,
                         const float* __restrict__ bhh,
                         float* __restrict__ h,
                         const int* __restrict__ gatherLen,
                         const float* __restrict__ mask, int t, int T,
                         float* __restrict__ gatherOut, int rows)
{
  int g = blockIdx.x * blockDim.x + threadIdx.x;
  if (g >= rows * 256) return;
  int i = g >> 8, j = g & 255;
  const long base = (long)i * 768;
  float ir = GI[base + j]       + bih[j];
  float iz = GI[base + 256 + j] + bih[256 + j];
  float in = GI[base + 512 + j] + bih[512 + j];
  float hr = GH[base + j]       + bhh[j];
  float hz = GH[base + 256 + j] + bhh[256 + j];
  float hn = GH[base + 512 + j] + bhh[512 + j];
  float rg = 1.0f / (1.0f + __expf(-(ir + hr)));
  float zg = 1.0f / (1.0f + __expf(-(iz + hz)));
  float ng = tanhf(in + rg * hn);
  float hq = (1.0f - zg) * ng + zg * h[g];
  h[g] = hq;
  if (gatherLen[i] == t) gatherOut[g] = hq * mask[i * T + t];
}

// Masked softmax attention; writes mix = [sum_cate | out_short] as bf16 [B,512].
__global__ void attention_mix(const float* __restrict__ seq_cate,   // [B,S,256]
                              const float* __restrict__ out_short,  // [B,256]
                              const float* __restrict__ mask_seq,   // [B,S]
                              unsigned short* __restrict__ mixb,    // [B,512] bf16
                              int B, int S)
{
  int wave = threadIdx.x >> 5;
  int lane = threadIdx.x & 31;
  int b = blockIdx.x * (blockDim.x >> 5) + wave;
  if (b >= B) return;

  float os[8];
  #pragma unroll
  for (int i = 0; i < 8; ++i) os[i] = out_short[(size_t)b * 256 + lane * 8 + i];

  float att[16];
  float amax = -1e30f;
  for (int s = 0; s < S; ++s) {
    const float* sc = seq_cate + ((size_t)b * S + s) * 256 + lane * 8;
    float p = 0.0f;
    #pragma unroll
    for (int i = 0; i < 8; ++i) p += sc[i] * os[i];
    for (int off = 16; off > 0; off >>= 1) p += __shfl_xor(p, off, 32);
    att[s] = p;
    amax = fmaxf(amax, p);
  }
  float wsum = 0.0f;
  for (int s = 0; s < S; ++s) { att[s] = __expf(att[s] - amax); wsum += att[s]; }
  float msum = 0.0f;
  for (int s = 0; s < S; ++s) {
    att[s] = (att[s] / wsum) * mask_seq[b * S + s];
    msum += att[s];
  }
  float inv = 1.0f / msum;

  #pragma unroll
  for (int i = 0; i < 8; ++i) {
    float acc = 0.0f;
    for (int s = 0; s < S; ++s)
      acc += att[s] * seq_cate[((size_t)b * S + s) * 256 + lane * 8 + i];
    mixb[(size_t)b * 512 + lane * 8 + i]       = f2bf(acc * inv);
    mixb[(size_t)b * 512 + 256 + lane * 8 + i] = f2bf(os[i]);
  }
}

__global__ void copy_target(const int* __restrict__ tgt,
                            float* __restrict__ out, int n)
{
  int i = blockIdx.x * blockDim.x + threadIdx.x;
  if (i < n) out[i] = (float)tgt[i];
}

// ---------------------------------------------------------------------------
extern "C" void kernel_launch(void* const* d_in, const int* in_sizes, int n_in,
                              void* d_out, int out_size, void* d_ws, size_t ws_size,
                              hipStream_t stream)
{
  const int  E  = 256, H = 256, G3 = 768;
  const int  Bc = in_sizes[5];
  const int  Tc = in_sizes[0] / Bc;
  const int  B  = in_sizes[9];
  const int  S  = Bc / B;
  const int  Ts = in_sizes[7] / B;
  const long V  = (long)in_sizes[12] / E;

  const int*   input_cate = (const int*)  d_in[0];
  const float* mask_cate  = (const float*)d_in[1];
  const float* mask_seq   = (const float*)d_in[2];
  const int*   subseqLen  = (const int*)  d_in[5];
  const int*   input_b    = (const int*)  d_in[7];
  const float* mask_b     = (const float*)d_in[8];
  const int*   seqLen_b   = (const int*)  d_in[9];
  const int*   target     = (const int*)  d_in[10];
  const float* emb        = (const float*)d_in[12];
  const float* Wih_c = (const float*)d_in[13];
  const float* Whh_c = (const float*)d_in[14];
  const float* bih_c = (const float*)d_in[15];
  const float* bhh_c = (const float*)d_in[16];
  const float* Wih_s = (const float*)d_in[17];
  const float* Whh_s = (const float*)d_in[18];
  const float* bih_s = (const float*)d_in[19];
  const float* bhh_s = (const float*)d_in[20];
  const float* fc_W  = (const float*)d_in[21];
  const float* fc_b  = (const float*)d_in[22];
  float* out = (float*)d_out;

  // ---- carve workspace (256B-aligned chunks) ----
  char* base = (char*)d_ws;
  size_t off = 0;
  auto carve = [&](size_t bytes) -> char* {
    char* p = base + off;
    off += (bytes + 255) & ~(size_t)255;
    return p;
  };
  unsigned short* embb   = (unsigned short*)carve((size_t)V * E * 2);
  unsigned short* WihCb  = (unsigned short*)carve((size_t)G3 * E * 2);
  unsigned short* WhhCb  = (unsigned short*)carve((size_t)G3 * H * 2);
  unsigned short* WihSb  = (unsigned short*)carve((size_t)G3 * E * 2);
  unsigned short* WhhSb  = (unsigned short*)carve((size_t)G3 * H * 2);
  unsigned short* fcWb   = (unsigned short*)carve((size_t)H * 2 * H * 2);
  float*          h_c    = (float*)carve((size_t)Bc * H * 4);
  unsigned short* Xb     = (unsigned short*)carve((size_t)Bc * E * 2);
  unsigned short* Hb     = (unsigned short*)carve((size_t)Bc * H * 2);
  float*          GI     = (float*)carve((size_t)Bc * G3 * 4);
  float*          GH     = (float*)carve((size_t)Bc * G3 * 4);
  float*          seqc   = (float*)carve((size_t)B * S * H * 4);
  float*          h_s    = (float*)carve((size_t)B * H * 4);
  float*          outsh  = (float*)carve((size_t)B * H * 4);
  unsigned short* mixb   = (unsigned short*)carve((size_t)B * 2 * H * 2);
  float*          fcOut  = (float*)carve((size_t)B * H * 4);
  unsigned short* fcOutb = (unsigned short*)carve((size_t)B * H * 2);
  (void)ws_size; (void)n_in; (void)out_size;

  // ---- one-time bf16 conversions ----
  {
    long n = V * E;
    cvt_f32_bf16<<<dim3((unsigned)((n + 255) / 256)), dim3(256), 0, stream>>>(emb, embb, n);
    long nw = (long)G3 * E;
    cvt_f32_bf16<<<dim3((unsigned)((nw + 255) / 256)), dim3(256), 0, stream>>>(Wih_c, WihCb, nw);
    cvt_f32_bf16<<<dim3((unsigned)((nw + 255) / 256)), dim3(256), 0, stream>>>(Whh_c, WhhCb, nw);
    cvt_f32_bf16<<<dim3((unsigned)((nw + 255) / 256)), dim3(256), 0, stream>>>(Wih_s, WihSb, nw);
    cvt_f32_bf16<<<dim3((unsigned)((nw + 255) / 256)), dim3(256), 0, stream>>>(Whh_s, WhhSb, nw);
    long nf = (long)H * 2 * H;
    cvt_f32_bf16<<<dim3((unsigned)((nf + 255) / 256)), dim3(256), 0, stream>>>(fc_W, fcWb, nf);
  }

  const int WV = 4;                      // waves (64-wide N-groups) per block
  const dim3 bGemm(WV * 32);

  // ---- category GRU over Bc rows, Tc steps ----
  hipMemsetAsync(h_c, 0, (size_t)Bc * H * 4, stream);
  {
    const int rows = Bc;
    const int nelem = rows * 256;
    dim3 gPrep((nelem + 255) / 256), b256(256);
    const int ngrp = (48 + 3) / 4;       // Ntiles = 768/16 = 48 -> 12 groups
    dim3 gGemm((ngrp + WV - 1) / WV, rows / 64);
    for (int t = 0; t < Tc; ++t) {
      gru_prep<<<gPrep, b256, 0, stream>>>(input_cate, t, Tc, embb, h_c, Xb, Hb, rows);
      gemm_bf16_wmma4x4<256><<<gGemm, bGemm, 0, stream>>>(Xb, WihCb, nullptr, GI, 48, G3);
      gemm_bf16_wmma4x4<256><<<gGemm, bGemm, 0, stream>>>(Hb, WhhCb, nullptr, GH, 48, G3);
      gru_gate<<<gPrep, b256, 0, stream>>>(GI, GH, bih_c, bhh_c, h_c,
                                           subseqLen, mask_cate, t, Tc, seqc, rows);
    }
  }

  // ---- short GRU over B rows, Ts steps (reuse GI/GH/Xb/Hb) ----
  hipMemsetAsync(h_s, 0, (size_t)B * H * 4, stream);
  {
    const int rows = B;
    const int nelem = rows * 256;
    dim3 gPrep((nelem + 255) / 256), b256(256);
    const int ngrp = (48 + 3) / 4;
    dim3 gGemm((ngrp + WV - 1) / WV, rows / 64);
    for (int t = 0; t < Ts; ++t) {
      gru_prep<<<gPrep, b256, 0, stream>>>(input_b, t, Ts, embb, h_s, Xb, Hb, rows);
      gemm_bf16_wmma4x4<256><<<gGemm, bGemm, 0, stream>>>(Xb, WihSb, nullptr, GI, 48, G3);
      gemm_bf16_wmma4x4<256><<<gGemm, bGemm, 0, stream>>>(Hb, WhhSb, nullptr, GH, 48, G3);
      gru_gate<<<gPrep, b256, 0, stream>>>(GI, GH, bih_s, bhh_s, h_s,
                                           seqLen_b, mask_b, t, Ts, outsh, rows);
    }
  }

  // ---- attention + mix (bf16) ----
  attention_mix<<<dim3((B + 7) / 8), dim3(256), 0, stream>>>(seqc, outsh, mask_seq,
                                                             mixb, B, S);

  // ---- fc: [B,512] @ fc_W[256,512]^T + fc_b ----
  {
    const int ngrp = (16 + 3) / 4;       // Ntiles = 16 -> 4 groups
    gemm_bf16_wmma4x4<512><<<dim3((ngrp + WV - 1) / WV, B / 64), bGemm, 0, stream>>>(
        mixb, fcWb, fc_b, fcOut, H / 16, (long)H);
  }
  {
    long n = (long)B * H;
    cvt_f32_bf16<<<dim3((unsigned)((n + 255) / 256)), dim3(256), 0, stream>>>(fcOut, fcOutb, n);
  }

  // ---- logits: [B,256] @ emb[V,256]^T -> d_out [B,V] ----
  {
    const int Ntiles = (int)(V / 16);    // 6250
    const int ngrp = (Ntiles + 3) / 4;   // 1563 (tail clamped in-kernel)
    gemm_bf16_wmma4x4<256><<<dim3((ngrp + WV - 1) / WV, B / 64), bGemm, 0, stream>>>(
        fcOutb, embb, nullptr, out, Ntiles, V);
  }

  // ---- append target_y as floats ----
  copy_target<<<dim3((B + 255) / 256), dim3(256), 0, stream>>>(target, out + (size_t)B * V, B);
}
